// Unet_2ring_51505247813776
// MI455X (gfx1250) — compile-verified, matrix operand order checked by reading, not run on hardware
//
#include <hip/hip_runtime.h>
#include <hip/hip_bf16.h>

typedef __bf16 bf16;
typedef __attribute__((ext_vector_type(16))) __bf16 v16bf;
typedef __attribute__((ext_vector_type(8)))  float  v8f;
typedef __attribute__((ext_vector_type(4)))  float  v4f;

// fp32 -> bf16 round-to-nearest-even via bit math
static __device__ __forceinline__ bf16 f2bf(float f) {
  union { float f; unsigned u; } c; c.f = f;
  unsigned r = (c.u + 0x7FFFu + ((c.u >> 16) & 1u)) >> 16;
  unsigned short s = (unsigned short)r;
  bf16 h;
  __builtin_memcpy(&h, &s, 2);
  return h;
}

// Pack W (coOut x K, fp32 row-major) into WMMA B-fragment-major bf16:
// Wp[kt][nt][lane][e] contiguous; e-ordering == ISA 16-bit B operand striping:
//   e<8 : k = kt*32 + (lane>>4)*8 + e ; e>=8 : k = kt*32 + 16 + (lane>>4)*8 + (e-8)
//   n = nt*16 + (lane&15)     (zero-filled for k>=K or n>=coOut)
// nTilesPad is a multiple of 4 so the GEMM inner loop needs no column guard.
__global__ __launch_bounds__(256) void k_pack_wt(const float* __restrict__ W,
                                                 bf16* __restrict__ Wp,
                                                 int K, int Kpad, int coOut, int nTilesPad) {
  long i = (long)blockIdx.x * 256 + threadIdx.x;
  long tot = (long)(Kpad >> 5) * nTilesPad * 512;
  if (i >= tot) return;
  int  e    = (int)(i & 15);
  int  L    = (int)((i >> 4) & 31);
  long tile = i >> 9;
  int  nt   = (int)(tile % nTilesPad);
  int  kt   = (int)(tile / nTilesPad);
  int  half = L >> 4, lm = L & 15;
  int  k = kt * 32 + ((e < 8) ? (half * 8 + e) : (16 + half * 8 + (e - 8)));
  int  n = nt * 16 + lm;
  float v = (k < K && n < coOut) ? W[(size_t)n * K + k] : 0.f;
  Wp[i] = f2bf(v);
}

// Gather-fused WMMA GEMM:  Y[N x coOut] = gather(H)[N x K] @ W^T + bias
// MODE 0: dense A (K == C, K%32==0), vector loads
// MODE 1: 19-neighbor gather, C%32==0 (K=19C, K%32==0), vector loads
// MODE 2: 19-neighbor gather, any C (scalar predicated; only down1.conv1, C=2)
// One wave32 owns a 16x64 output strip: 4 WMMA accumulators, double-buffered k-loop.
template <int MODE>
__global__ __launch_bounds__(256) void k_gemm(
    const float* __restrict__ H, const int* __restrict__ neigh,
    const bf16* __restrict__ Wp, const float* __restrict__ bias,
    float* __restrict__ Y,
    int N, int C, int log2C, int K, int Kpad, int nTilesPad, int coOut,
    int mTiles, int nGroups) {
  int lane = threadIdx.x & 31;
  int wave = (int)(blockIdx.x * (blockDim.x >> 5) + (threadIdx.x >> 5));
  if (wave >= mTiles * nGroups) return;          // uniform whole-wave exit
  // force provably-scalar tile coordinates (kills exec-mask branching)
  int mTile = __builtin_amdgcn_readfirstlane(wave % mTiles);
  int nG    = __builtin_amdgcn_readfirstlane(wave / mTiles);
  int half  = lane >> 4;
  int lm    = lane & 15;
  int row   = mTile * 16 + lm;
  int rowc  = (row < N) ? row : (N - 1);         // clamp: EXEC all-ones through WMMA

  v8f acc[4];
#pragma unroll
  for (int t = 0; t < 4; ++t)
#pragma unroll
    for (int e = 0; e < 8; ++e) acc[t][e] = 0.f;

  const bf16* bp = Wp + ((size_t)nG * 4) * 512 + (size_t)lane * 16;
  const int kTiles = Kpad >> 5;

  auto loadA = [&](int kt, v16bf& a) {
    int kc = kt * 32;
    if constexpr (MODE == 2) {
#pragma unroll
      for (int e = 0; e < 16; ++e) {
        int k = kc + ((e < 8) ? (half * 8 + e) : (16 + half * 8 + (e - 8)));
        float v = 0.f;
        if (k < K) {
          int j = k >> log2C, c = k & (C - 1);
          v = H[(size_t)neigh[(size_t)rowc * 19 + j] * C + c];
        }
        a[e] = f2bf(v);
      }
    } else {
      int k0 = kc + half * 8;        // 8 consecutive k
      int k1 = k0 + 16;              // 8 consecutive k
      const float *p0, *p1;
      if constexpr (MODE == 1) {
        int j0 = k0 >> log2C, c0 = k0 & (C - 1);
        int j1 = k1 >> log2C, c1 = k1 & (C - 1);
        p0 = H + (size_t)neigh[(size_t)rowc * 19 + j0] * C + c0;
        p1 = H + (size_t)neigh[(size_t)rowc * 19 + j1] * C + c1;
      } else {
        p0 = H + (size_t)rowc * C + k0;
        p1 = H + (size_t)rowc * C + k1;
      }
      v4f f0 = *(const v4f*)p0;
      v4f f1 = *(const v4f*)(p0 + 4);
      v4f f2 = *(const v4f*)p1;
      v4f f3 = *(const v4f*)(p1 + 4);
#pragma unroll
      for (int e = 0; e < 4; ++e) {
        a[e]      = f2bf(f0[e]);
        a[e + 4]  = f2bf(f1[e]);
        a[e + 8]  = f2bf(f2[e]);
        a[e + 12] = f2bf(f3[e]);
      }
    }
  };
  auto loadB = [&](int kt, v16bf* b) {
    const bf16* bk = bp + (size_t)kt * nTilesPad * 512;
#pragma unroll
    for (int t = 0; t < 4; ++t)
      b[t] = *(const v16bf*)(bk + (size_t)t * 512);   // 32B contiguous: 2x b128
  };

  // even/odd double-buffered pipeline: loads of chunk kt+1 overlap WMMAs of kt
  v16bf a0, a1, b0[4], b1[4];
  loadA(0, a0);
  loadB(0, b0);
  for (int kt = 0; kt < kTiles; kt += 2) {
    bool has1 = (kt + 1 < kTiles);
    if (has1) {
      loadA(kt + 1, a1);
      loadB(kt + 1, b1);
      if (kt + 2 < kTiles)
        __builtin_prefetch(bp + (size_t)(kt + 2) * nTilesPad * 512, 0, 1);
    }
#pragma unroll
    for (int t = 0; t < 4; ++t)
      acc[t] = __builtin_amdgcn_wmma_f32_16x16x32_bf16(
          false, a0, false, b0[t], (short)0, acc[t], false, false);
    if (has1) {
      if (kt + 2 < kTiles) {
        loadA(kt + 2, a0);
        loadB(kt + 2, b0);
      }
#pragma unroll
      for (int t = 0; t < 4; ++t)
        acc[t] = __builtin_amdgcn_wmma_f32_16x16x32_bf16(
            false, a1, false, b1[t], (short)0, acc[t], false, false);
    }
  }

  // D layout: lanes 0-15 -> M = r, lanes 16-31 -> M = r+8; N = lm
#pragma unroll
  for (int t = 0; t < 4; ++t) {
    int col = (nG * 4 + t) * 16 + lm;
#pragma unroll
    for (int r = 0; r < 8; ++r) {
      int rr = mTile * 16 + half * 8 + r;
      if (rr < N && col < coOut)
        Y[(size_t)rr * coOut + col] = acc[t][r] + bias[col];
    }
  }
}

// Per-column mean + biased variance over N rows. grid = co, block = 256.
__global__ __launch_bounds__(256) void k_colstats(const float* __restrict__ H,
                                                  float* __restrict__ stats,
                                                  int N, int co) {
  __shared__ float sh[256], sh2[256];
  int c = blockIdx.x;
  float s = 0.f, s2 = 0.f;
  for (int i = threadIdx.x; i < N; i += 256) {
    float v = H[(size_t)i * co + c];
    s += v; s2 += v * v;
  }
  sh[threadIdx.x] = s; sh2[threadIdx.x] = s2;
  __syncthreads();
  for (int o = 128; o > 0; o >>= 1) {
    if ((int)threadIdx.x < o) {
      sh[threadIdx.x]  += sh[threadIdx.x + o];
      sh2[threadIdx.x] += sh2[threadIdx.x + o];
    }
    __syncthreads();
  }
  if (threadIdx.x == 0) {
    float mu = sh[0] / (float)N;
    stats[c] = mu;
    stats[co + c] = sh2[0] / (float)N - mu * mu;
  }
}

// In-place BatchNorm (batch stats) + LeakyReLU(0.2)
__global__ __launch_bounds__(256) void k_bnleaky(float* __restrict__ H,
                                                 const float* __restrict__ stats,
                                                 const float* __restrict__ g,
                                                 const float* __restrict__ b,
                                                 int N, int co) {
  long i = (long)blockIdx.x * 256 + threadIdx.x;
  if (i >= (long)N * co) return;
  int c = (int)(i % co);
  float mu = stats[c], var = stats[co + c];
  float y = (H[i] - mu) * rsqrtf(var + 1e-5f) * g[c] + b[c];
  H[i] = (y >= 0.f) ? y : 0.2f * y;
}

// Mean pool over 7 gathered fine-level vertices
__global__ __launch_bounds__(256) void k_pool(const float* __restrict__ Hin,
                                              const int* __restrict__ pidx,
                                              float* __restrict__ out, int Nout, int C) {
  long i = (long)blockIdx.x * 256 + threadIdx.x;
  if (i >= (long)Nout * C) return;
  int v = (int)(i / C), c = (int)(i % C);
  float s = 0.f;
#pragma unroll
  for (int j = 0; j < 7; ++j)
    s += Hin[(size_t)pidx[(size_t)v * 7 + j] * C + c];
  out[i] = s * (1.f / 7.f);
}

// upconv index-select (+pair mean) and channel-concat with skip
__global__ __launch_bounds__(256) void k_upcat(const float* __restrict__ Yup,
                                               const int* __restrict__ top,
                                               const int* __restrict__ dwn,
                                               const float* __restrict__ skip,
                                               float* __restrict__ out,
                                               int Nnew, int raw, int co) {
  long i = (long)blockIdx.x * 256 + threadIdx.x;
  if (i >= (long)Nnew * co) return;
  int v = (int)(i / co), c = (int)(i % co);
  float left;
  if (v < raw) {
    left = Yup[(size_t)top[v] * co + c];
  } else {
    int t = v - raw;
    left = 0.5f * (Yup[(size_t)dwn[2 * t] * co + c] + Yup[(size_t)dwn[2 * t + 1] * co + c]);
  }
  out[(size_t)v * (2 * co) + c]      = left;
  out[(size_t)v * (2 * co) + co + c] = skip[(size_t)v * co + c];
}

extern "C" void kernel_launch(void* const* d_in, const int* in_sizes, int n_in,
                              void* d_out, int out_size, void* d_ws, size_t ws_size,
                              hipStream_t stream) {
  (void)in_sizes; (void)n_in; (void)out_size; (void)ws_size;
  // input indices per setup_inputs() dict order:
  // 0: x; down i: base 1+(i-1)*8 = {W1,b1,g1,be1,W2,b2,g2,be2}
  // up j: base 41+(j-1)*10 = {uW,ub,W1,b1,g1,be1,W2,b2,g2,be2}
  // 81: out.W, 82: out.b; 83..87: n1..n5; 88..91: p2..p5; 92..99: ut1,ud1..ut4,ud4
  auto F = [&](int i) { return (const float*)d_in[i]; };
  auto I = [&](int i) { return (const int*)d_in[i]; };

  size_t off = 0;
  auto alloc = [&](size_t bytes) -> void* {
    void* p = (char*)d_ws + off;
    off = (off + bytes + 255) & ~(size_t)255;
    return p;
  };
  float* x2f   = (float*)alloc(40962ull * 64   * 4);
  float* x3f   = (float*)alloc(10242ull * 128  * 4);
  float* x4f   = (float*)alloc(2562ull  * 256  * 4);
  float* x5f   = (float*)alloc(642ull   * 512  * 4);
  float* x6f   = (float*)alloc(162ull   * 1024 * 4);
  float* poolb = (float*)alloc(10242ull * 64   * 4);
  float* t1    = (float*)alloc(40962ull * 64   * 4);
  float* ybuf  = (float*)alloc(10242ull * 448  * 4);
  float* catb  = (float*)alloc(40962ull * 128  * 4);
  float* hA    = (float*)alloc(40962ull * 64   * 4);
  float* hB    = (float*)alloc(40962ull * 64   * 4);
  bf16*  wbuf  = (bf16*) alloc(19456ull * 1024 * 2);   // largest packed Wt (down5 W2)
  float* statb = (float*)alloc(2048ull * 4);

  auto ru = [](int x, int m) { return (x + m - 1) / m * m; };

  auto gemm = [&](const float* Hsrc, const int* neigh, const float* W, const float* bias,
                  float* Y, int N, int C, int K, int coOut) {
    int Kp        = ru(K, 32);
    int nTilesPad = ru(ru(coOut, 16) / 16, 4);   // multiple of 4: no guard in k-loop
    long tot      = (long)(Kp >> 5) * nTilesPad * 512;
    k_pack_wt<<<dim3((unsigned)((tot + 255) / 256)), dim3(256), 0, stream>>>(
        W, wbuf, K, Kp, coOut, nTilesPad);
    int mT = (N + 15) / 16;
    int nG = nTilesPad / 4;
    long waves = (long)mT * nG;
    int l2c = 0; while ((1 << l2c) < C) ++l2c;
    dim3 grid((unsigned)((waves + 7) / 8)), blk(256);
    if (!neigh)
      k_gemm<0><<<grid, blk, 0, stream>>>(Hsrc, neigh, wbuf, bias, Y, N, C, l2c, K, Kp,
                                          nTilesPad, coOut, mT, nG);
    else if ((C & 31) == 0)
      k_gemm<1><<<grid, blk, 0, stream>>>(Hsrc, neigh, wbuf, bias, Y, N, C, l2c, K, Kp,
                                          nTilesPad, coOut, mT, nG);
    else
      k_gemm<2><<<grid, blk, 0, stream>>>(Hsrc, neigh, wbuf, bias, Y, N, C, l2c, K, Kp,
                                          nTilesPad, coOut, mT, nG);
  };

  auto dconv = [&](const float* Hin, int N, int ci, const int* neigh, int pb, int co,
                   float* out) {
    gemm(Hin, neigh, F(pb + 0), F(pb + 1), t1, N, ci, 19 * ci, co);
    k_colstats<<<dim3((unsigned)co), dim3(256), 0, stream>>>(t1, statb, N, co);
    long tot = (long)N * co;
    k_bnleaky<<<dim3((unsigned)((tot + 255) / 256)), dim3(256), 0, stream>>>(
        t1, statb, F(pb + 2), F(pb + 3), N, co);
    gemm(t1, neigh, F(pb + 4), F(pb + 5), out, N, co, 19 * co, co);
    k_colstats<<<dim3((unsigned)co), dim3(256), 0, stream>>>(out, statb, N, co);
    k_bnleaky<<<dim3((unsigned)((tot + 255) / 256)), dim3(256), 0, stream>>>(
        out, statb, F(pb + 6), F(pb + 7), N, co);
  };

  auto pool = [&](const float* Hin, const int* pidx, float* out, int Nout, int C) {
    long tot = (long)Nout * C;
    k_pool<<<dim3((unsigned)((tot + 255) / 256)), dim3(256), 0, stream>>>(Hin, pidx, out, Nout, C);
  };

  // ---------------- encoder ----------------
  dconv(F(0),  40962, 2,   I(83), 1,  64,   x2f);
  pool(x2f, I(88), poolb, 10242, 64);
  dconv(poolb, 10242, 64,  I(84), 9,  128,  x3f);
  pool(x3f, I(89), poolb, 2562, 128);
  dconv(poolb, 2562,  128, I(85), 17, 256,  x4f);
  pool(x4f, I(90), poolb, 642, 256);
  dconv(poolb, 642,   256, I(86), 25, 512,  x5f);
  pool(x5f, I(91), poolb, 162, 512);
  dconv(poolb, 162,   512, I(87), 33, 1024, x6f);

  // ---------------- decoder ----------------
  auto upblk = [&](const float* h, int Nraw, int ci, const float* skip, int Nnew,
                   const int* top, const int* dwn, const int* neigh, int pb, float* out) {
    int co = ci >> 1;
    gemm(h, nullptr, F(pb + 0), F(pb + 1), ybuf, Nraw, ci, ci, 7 * co);  // (raw,7co)=(raw*7,co)
    long tot = (long)Nnew * co;
    k_upcat<<<dim3((unsigned)((tot + 255) / 256)), dim3(256), 0, stream>>>(
        ybuf, top, dwn, skip, catb, Nnew, Nraw, co);
    dconv(catb, Nnew, ci, neigh, pb + 2, co, out);
  };

  upblk(x6f, 162,   1024, x5f, 642,   I(92), I(93), I(86), 41, hA);  // -> (642, 512)
  upblk(hA,  642,   512,  x4f, 2562,  I(94), I(95), I(85), 51, hB);  // -> (2562, 256)
  upblk(hB,  2562,  256,  x3f, 10242, I(96), I(97), I(84), 61, hA);  // -> (10242, 128)
  upblk(hA,  10242, 128,  x2f, 40962, I(98), I(99), I(83), 71, hB);  // -> (40962, 64)

  // ---------------- output head ----------------
  gemm(hB, nullptr, F(81), F(82), (float*)d_out, 40962, 64, 64, 36);
}
